// RecurrentDecoder_33363305955846
// MI455X (gfx1250) — matrix operand order repaired and checked by reading, not compiled, not run on hardware
//
#include <hip/hip_runtime.h>
#include <cstdint>

// ---------------- problem constants ----------------
#define B_   16
#define T_   256
#define TM   255          // T-1
#define LAT  512
#define U_   1024
#define G3   3072         // 3*U
#define V_   32000
#define E_   256
#define MPAD 4096         // 4080 rows padded to 256 16-row tiles

// ---------------- workspace layout (bytes, 256-aligned) ----------------
#define OFF_H0     0u                         // 16*1024 f32          = 65,536
#define OFF_XPROJ  65536u                     // 4080*3072 f32        = 50,135,040
#define OFF_YBF    50200576u                  // 4096*1024 bf16 (pad) = 8,388,608
#define OFF_RKT    58589184u                  // 3072*1024 bf16 (T)   = 6,291,456
#define OFF_GKT    64880640u                  // 3072*256  bf16 (T)   = 1,572,864
#define OFF_OWT    66453504u                  // 32000*1024 bf16 (T)  = 65,536,000
#define OFF_HBF    131989504u                 // 16*1024 bf16 h state = 32,768
#define OFF_RPA    132022272u                 // 16*3072 f32 partial  = 196,608
#define OFF_RPB    132218880u                 // 16*3072 f32 partial  = 196,608
#define OFF_SYNC   132415488u                 // grid-sync counter
// total ~132.4 MB

#define NBLK 16                               // cooperative blocks for the GRU

typedef unsigned short u16;  // bf16 bit-carrier across host/device ABI
typedef __attribute__((ext_vector_type(16))) __bf16 v16bf;
typedef __attribute__((ext_vector_type(8)))  float  v8f;

union ABf16 { v16bf v; uint4 q[2]; };

// D = A x B + C, bf16 inputs, f32 accum
__device__ static inline v8f wmma_bf16(v16bf a, v16bf b, v8f c) {
  return __builtin_amdgcn_wmma_f32_16x16x32_bf16(
      /*neg_a=*/false, a, /*neg_b=*/false, b,
      /*c_mod=*/(short)0, c, /*reuse_a=*/false, /*reuse_b=*/false);
}

// A-tile (16x32, MxK) from row-major f32 source; `row` = &src[m_row * ld].
// ISA 16-bit A layout: lane m = L&15; element i -> K = 16*(i>>3) + 8*(L>>4) + (i&7)
__device__ static inline v16bf a_from_f32row(const float* __restrict__ row, int k0, int lane) {
  int kb = (lane >> 4) * 8;
  v16bf a;
#pragma unroll
  for (int g = 0; g < 2; ++g) {
    int kbase = k0 + g * 16 + kb;
#pragma unroll
    for (int j = 0; j < 8; ++j) a[g * 8 + j] = (__bf16)row[kbase + j];
  }
  return a;
}

// A-tile from row-major bf16 source: two 16B loads per lane.
__device__ static inline v16bf a_from_bf16row(const __bf16* __restrict__ row, int k0, int lane) {
  int kb = (lane >> 4) * 8;
  ABf16 r;
  r.q[0] = *(const uint4*)(row + k0 + kb);        // K = kb .. kb+7
  r.q[1] = *(const uint4*)(row + k0 + 16 + kb);   // K = 16+kb .. 16+kb+7
  return r.v;
}

// B-tile (32x16, KxN) from bf16 column-major (transposed) source;
// `colT` = &srcT[n_col * K]. ISA 16-bit B layout: lane n = L&15;
// element i -> K = 16*(i>>3) + 8*(L>>4) + (i&7) -> two 16B loads.
__device__ static inline v16bf b_from_bf16T(const __bf16* __restrict__ colT, int k0, int lane) {
  int kb = (lane >> 4) * 8;
  ABf16 r;
  r.q[0] = *(const uint4*)(colT + k0 + kb);
  r.q[1] = *(const uint4*)(colT + k0 + 16 + kb);
  return r.v;
}

// B-tile from row-major f32 weights (strided, converted on the fly).
__device__ static inline v16bf b_from_f32rm(const float* __restrict__ src, int ldn,
                                            int k0, int n, int lane) {
  int kb = (lane >> 4) * 8;
  v16bf b;
#pragma unroll
  for (int i = 0; i < 16; ++i)
    b[i] = (__bf16)src[(size_t)(k0 + kb + (i & 7) + ((i >> 3) << 4)) * ldn + n];
  return b;
}

// Store 16x16 f32 D tile + bias(n). C/D layout: lane n = L&15, VGPR j -> m = 8*(L>>4)+j.
__device__ static inline void store_d(float* __restrict__ dst, int ld, v8f c,
                                      float bias, int lane) {
  int n  = lane & 15;
  int mb = (lane >> 4) * 8;
#pragma unroll
  for (int j = 0; j < 8; ++j) dst[(size_t)(mb + j) * ld + n] = c[j] + bias;
}

// ---------------- init: zero ybf pad tile + reset grid-sync counter ----------------
__global__ void k_init(u16* __restrict__ ybf_pad, int* __restrict__ sync) {
  int idx = blockIdx.x * blockDim.x + threadIdx.x;
  if (idx < 16 * U_) ybf_pad[idx] = 0;       // rows 4080..4095
  if (idx == 0) *sync = 0;
}

// ---------------- one-time convert + transpose: f32 (K x N) -> bf16 (N x K) ----------------
__global__ void cvt_transpose(const float* __restrict__ src, u16* __restrict__ dst_,
                              int K, int N) {
  __bf16* dst = (__bf16*)dst_;
  int idx = blockIdx.x * blockDim.x + threadIdx.x;
  if (idx >= K * N) return;
  int k = idx / N, n = idx - k * N;
  dst[(size_t)n * K + k] = (__bf16)src[idx];
}

// ---------------- h0 = latent @ dense_W + dense_b  (16 x 512 x 1024) ----------------
__global__ void k_h0(const float* __restrict__ latent, const float* __restrict__ dW,
                     const float* __restrict__ db, float* __restrict__ h0) {
  int wid  = (blockIdx.x * blockDim.x + threadIdx.x) >> 5;
  int lane = threadIdx.x & 31;
  if (wid >= U_ / 16) return;                  // wave-uniform guard
  int n0 = wid * 16;
  const float* arow = latent + (size_t)(lane & 15) * LAT;
  v8f c = {0, 0, 0, 0, 0, 0, 0, 0};
  for (int k0 = 0; k0 < LAT; k0 += 32) {
    v16bf a = a_from_f32row(arow, k0, lane);
    v16bf b = b_from_f32rm(dW, U_, k0, n0 + (lane & 15), lane);
    c = wmma_bf16(a, b, c);
  }
  store_d(h0 + n0, U_, c, db[n0 + (lane & 15)], lane);
}

// ---------------- x_proj = emb[teacher] @ gru_kernel + bias0  (4080 x 256 x 3072) ----------------
__global__ void k_xproj(const long long* __restrict__ tokens,   // true_outputs (B,T) int64
                        const float* __restrict__ emb,
                        const u16* __restrict__ gkT_,           // (3072 x 256) bf16
                        const float* __restrict__ gbias,        // gru_bias row 0
                        float* __restrict__ xproj) {
  const __bf16* gkT = (const __bf16*)gkT_;
  int wid  = (blockIdx.x * blockDim.x + threadIdx.x) >> 5;
  int lane = threadIdx.x & 31;
  const int MT = (B_ * TM) / 16;   // 255
  const int NT = G3 / 16;          // 192
  if (wid >= MT * NT) return;
  int mt = wid / NT, nt = wid - mt * NT;
  int r = mt * 16 + (lane & 15);
  int b = r / TM, t = r - b * TM;
  long long tok = tokens[b * T_ + t];
  const float*  arow = emb + (size_t)tok * E_;
  int n0 = nt * 16;
  const __bf16* bcol = gkT + (size_t)(n0 + (lane & 15)) * E_;
  v8f c = {0, 0, 0, 0, 0, 0, 0, 0};
#pragma unroll
  for (int k0 = 0; k0 < E_; k0 += 32) {
    v16bf a  = a_from_f32row(arow, k0, lane);
    v16bf bm = b_from_bf16T(bcol, k0, lane);
    c = wmma_bf16(a, bm, c);
  }
  store_d(xproj + (size_t)mt * 16 * G3 + n0, G3, c, gbias[n0 + (lane & 15)], lane);
}

// ---------------- cooperative grid barrier (16 resident blocks) ----------------
__device__ static inline void grid_sync(int* ctr, int& target) {
  target += NBLK;
  __threadfence();
  __syncthreads();
  if (threadIdx.x == 0) {
    __hip_atomic_fetch_add(ctr, 1, __ATOMIC_ACQ_REL, __HIP_MEMORY_SCOPE_AGENT);
    while (__hip_atomic_load(ctr, __ATOMIC_ACQUIRE, __HIP_MEMORY_SCOPE_AGENT) < target)
      __builtin_amdgcn_s_sleep(1);
  }
  __syncthreads();
  __threadfence();
}

// ---------------- sequential GRU across 16 WGPs with grid barrier ----------------
// stage 1: rp = h @ gru_rkernel + bias1; 192 n-tiles = 16 blocks x 12 tiles,
//          2 waves per tile split K in halves -> partial buffers rpA/rpB.
// stage 2: thread owns one (m, n): gates + masked h/y update in registers.
__global__ __launch_bounds__(1024) void k_gru(
    const float* __restrict__ h0,
    const float* __restrict__ xproj,
    const u16* __restrict__ rkT_,           // gru_rkernel^T bf16 (3072 x 1024)
    const float* __restrict__ gbias,        // gru_bias base; row1 at +G3
    const long long* __restrict__ tokens,
    u16* __restrict__ ybf_,                 // gru_out bf16 (4096 x 1024, padded)
    u16* __restrict__ hbf_,                 // h state bf16 (16 x 1024)
    float* __restrict__ rpA,
    float* __restrict__ rpB,
    int* __restrict__ sync) {
  const __bf16* rkT = (const __bf16*)rkT_;
  __bf16* ybf = (__bf16*)ybf_;
  __bf16* hbf = (__bf16*)hbf_;

  const int tid  = threadIdx.x;
  const int lane = tid & 31;
  const int wv   = tid >> 5;               // wave 0..31
  const int blk  = blockIdx.x;             // 0..15

  // stage-2 ownership: one (m, n) per thread; 16 blocks x 1024 = 16384 pairs
  const int m2 = tid >> 6;                 // 0..15
  const int n2 = blk * 64 + (tid & 63);    // 0..1023
  float hreg = h0[m2 * U_ + n2];
  float yreg = 0.f;
  hbf[m2 * U_ + n2] = (__bf16)hreg;

  int target = 0;
  grid_sync(sync, target);                 // initial h visible everywhere

  // stage-1 wave assignment (uniform per wave): 24 active waves
  const bool s1_active = (wv < 24);
  const int  tile  = wv >> 1;              // 0..11
  const int  khalf = wv & 1;               // K in [khalf*512, khalf*512+512)
  const int  n0    = (blk * 12 + tile) * 16;
  const int  kbeg  = khalf * 512, kend = kbeg + 512;
  float* rpOut = khalf ? rpB : rpA;

  for (int t = 0; t < TM; ++t) {
    if (s1_active) {
      const __bf16* arow = hbf + (size_t)(lane & 15) * U_;
      const __bf16* bcol = rkT + (size_t)(n0 + (lane & 15)) * U_;
      v8f c = {0, 0, 0, 0, 0, 0, 0, 0};
#pragma unroll 4
      for (int k0 = kbeg; k0 < kend; k0 += 32) {
        v16bf a  = a_from_bf16row(arow, k0, lane);
        v16bf bm = b_from_bf16T(bcol, k0, lane);
        c = wmma_bf16(a, bm, c);
      }
      float bias = khalf ? 0.f : gbias[G3 + n0 + (lane & 15)];
      int n = lane & 15, mb = (lane >> 4) * 8;
#pragma unroll
      for (int j = 0; j < 8; ++j) rpOut[(mb + j) * G3 + n0 + n] = c[j] + bias;
    }
    grid_sync(sync, target);

    // stage 2: gates for this thread's (m2, n2)
    {
      size_t xo = ((size_t)m2 * TM + t) * G3;
      float xz = xproj[xo + n2];
      float xr = xproj[xo + U_ + n2];
      float xh = xproj[xo + 2 * U_ + n2];
      int   ro = m2 * G3 + n2;
      float rz = rpA[ro] + rpB[ro];
      float rr = rpA[ro + U_] + rpB[ro + U_];
      float rh = rpA[ro + 2 * U_] + rpB[ro + 2 * U_];
      float z  = 1.f / (1.f + __expf(-(xz + rz)));
      float rg = 1.f / (1.f + __expf(-(xr + rr)));
      float hh = tanhf(xh + rg * rh);
      float hn = z * hreg + (1.f - z) * hh;
      bool  mk = tokens[m2 * T_ + t] != 0;
      hreg = mk ? hn : hreg;
      yreg = mk ? hn : yreg;
      ybf[((size_t)m2 * TM + t) * U_ + n2] = (__bf16)yreg;
      hbf[m2 * U_ + n2] = (__bf16)hreg;
    }
    grid_sync(sync, target);
  }
}

// ---------------- logits = y @ out_W + out_b  (4096pad x 1024 x 32000) ----------------
// 4 m-tiles x 4 n-tiles per wave: 16 accumulators, operand traffic 32 B/KFLOP.
__global__ void k_logits(const u16* __restrict__ ybf_,
                         const u16* __restrict__ owT_,     // out_W^T bf16 (32000 x 1024)
                         const float* __restrict__ ob,
                         float* __restrict__ out) {
  const __bf16* ybf = (const __bf16*)ybf_;
  const __bf16* owT = (const __bf16*)owT_;
  int wid  = (blockIdx.x * blockDim.x + threadIdx.x) >> 5;
  int lane = threadIdx.x & 31;
  const int MG = MPAD / 64;        // 64 groups of 4 m-tiles
  const int NG = V_ / 64;          // 500 groups of 4 n-tiles
  if (wid >= MG * NG) return;      // wave-uniform guard
  int mg = wid / NG, ng = wid - mg * NG;
  int m0t = mg * 4;
  int n0  = ng * 64;

  const __bf16* arow[4];
#pragma unroll
  for (int i = 0; i < 4; ++i)
    arow[i] = ybf + (size_t)((m0t + i) * 16 + (lane & 15)) * U_;
  const __bf16* bcol[4];
#pragma unroll
  for (int j = 0; j < 4; ++j)
    bcol[j] = owT + (size_t)(n0 + j * 16 + (lane & 15)) * U_;

  v8f c[16];
#pragma unroll
  for (int i = 0; i < 16; ++i) c[i] = (v8f){0, 0, 0, 0, 0, 0, 0, 0};

  for (int k0 = 0; k0 < U_; k0 += 32) {
    if (k0 + 64 < U_) __builtin_prefetch((const void*)(arow[0] + k0 + 64), 0, 0);
    v16bf a[4], b[4];
#pragma unroll
    for (int i = 0; i < 4; ++i) a[i] = a_from_bf16row(arow[i], k0, lane);
#pragma unroll
    for (int j = 0; j < 4; ++j) b[j] = b_from_bf16T(bcol[j], k0, lane);
#pragma unroll
    for (int i = 0; i < 4; ++i)
#pragma unroll
      for (int j = 0; j < 4; ++j)
        c[i * 4 + j] = wmma_bf16(a[i], b[j], c[i * 4 + j]);
  }

  int nb = lane & 15;
#pragma unroll
  for (int i = 0; i < 4; ++i) {
    int mt = m0t + i;
    if (mt >= 255) continue;       // pad tile: computed on zeros, never stored
    float* dst = out + (size_t)mt * 16 * V_ + n0;
#pragma unroll
    for (int j = 0; j < 4; ++j)
      store_d(dst + j * 16, V_, c[i * 4 + j], ob[n0 + j * 16 + nb], lane);
  }
}

// ---------------- host launcher ----------------
extern "C" void kernel_launch(void* const* d_in, const int* in_sizes, int n_in,
                              void* d_out, int out_size, void* d_ws, size_t ws_size,
                              hipStream_t stream) {
  const float*     latent = (const float*)d_in[0];
  const long long* tokens = (const long long*)d_in[1];   // jnp.int64
  const float*     emb    = (const float*)d_in[2];
  const float*     dW     = (const float*)d_in[3];
  const float*     db     = (const float*)d_in[4];
  const float*     gk     = (const float*)d_in[5];
  const float*     grk    = (const float*)d_in[6];
  const float*     gb     = (const float*)d_in[7];
  const float*     oW     = (const float*)d_in[8];
  const float*     ob     = (const float*)d_in[9];
  float* out = (float*)d_out;

  char* ws = (char*)d_ws;
  float* h0    = (float*)(ws + OFF_H0);
  float* xproj = (float*)(ws + OFF_XPROJ);
  u16*   ybf   = (u16*)(ws + OFF_YBF);
  u16*   rkT   = (u16*)(ws + OFF_RKT);
  u16*   gkT   = (u16*)(ws + OFF_GKT);
  u16*   owT   = (u16*)(ws + OFF_OWT);
  u16*   hbf   = (u16*)(ws + OFF_HBF);
  float* rpA   = (float*)(ws + OFF_RPA);
  float* rpB   = (float*)(ws + OFF_RPB);
  int*   sync  = (int*)(ws + OFF_SYNC);

  k_init<<<64, 256, 0, stream>>>(ybf + (size_t)4080 * U_, sync);

  // one-time bf16 transposed weight copies (B tiles become two 16B loads/lane)
  cvt_transpose<<<(E_ * G3 + 255) / 256, 256, 0, stream>>>(gk,  gkT, E_, G3);
  cvt_transpose<<<(U_ * G3 + 255) / 256, 256, 0, stream>>>(grk, rkT, U_, G3);
  cvt_transpose<<<(U_ * V_ + 255) / 256, 256, 0, stream>>>(oW,  owT, U_, V_);

  k_h0<<<8, 256, 0, stream>>>(latent, dW, db, h0);                       // 64 waves
  k_xproj<<<(255 * 192 + 7) / 8, 256, 0, stream>>>(tokens, emb, gkT, gb, xproj);
  k_gru<<<NBLK, 1024, 0, stream>>>(h0, xproj, rkT, gb, tokens,
                                   ybf, hbf, rpA, rpB, sync);            // 16 WGPs
  k_logits<<<(64 * 500) / 8, 256, 0, stream>>>(ybf, owT, ob, out);       // dominant GEMM
}
